// SelfAttentionModule_65481071406093
// MI455X (gfx1250) — matrix operand order repaired and checked by reading, not compiled, not run on hardware
//
#include <hip/hip_runtime.h>
#include <hip/hip_bf16.h>

#define B_  2
#define L_  2048
#define D_  1024
#define H_  16
#define HD_ 64

typedef __bf16 bf16_t;
typedef __attribute__((ext_vector_type(16))) __bf16 v16bf;
typedef __attribute__((ext_vector_type(8)))  __bf16 v8bf;
typedef __attribute__((ext_vector_type(4)))  __bf16 v4bf;
typedef __attribute__((ext_vector_type(8)))  float  v8f;
typedef __attribute__((ext_vector_type(4)))  float  v4f;

__device__ __forceinline__ v16bf cat8(v8bf lo, v8bf hi) {
    v16bf r;
#pragma unroll
    for (int i = 0; i < 8; ++i) { r[i] = lo[i]; r[i + 8] = hi[i]; }
    return r;
}

// ---------------------------------------------------------------------------
// Kernel 1: fp32 -> bf16 conversion of x and the three weight matrices.
// ---------------------------------------------------------------------------
__global__ void cvt_kernel(const float* __restrict__ x,  const float* __restrict__ wq,
                           const float* __restrict__ wk, const float* __restrict__ wv,
                           bf16_t* __restrict__ xbf, bf16_t* __restrict__ wbf) {
    const int nx = B_ * L_ * D_;      // 4194304
    const int nw = D_ * D_;           // 1048576
    const int total = nx + 3 * nw;
    int i = (blockIdx.x * blockDim.x + threadIdx.x) * 4;
    if (i >= total) return;
    const float* src;
    bf16_t* dst;
    if (i < nx) { src = x + i; dst = xbf + i; }
    else {
        int j = i - nx;
        int m = j / nw;
        int o = j - m * nw;
        src = (m == 0 ? wq : (m == 1 ? wk : wv)) + o;
        dst = wbf + j;
    }
    v4f f = *(const v4f*)src;
    v4bf o4;
#pragma unroll
    for (int k = 0; k < 4; ++k) o4[k] = (bf16_t)f[k];
    *(v4bf*)dst = o4;
}

// ---------------------------------------------------------------------------
// Kernel 2: fused QKV projection  y = x @ W^T  with v_wmma_f32_16x16x32_bf16.
// One wave computes a 16(M) x 64(N) tile of one of {Q,K,V}.
// Q,K written row-major (B,H,L,HD); V written transposed (B,H,HD,L).
// ---------------------------------------------------------------------------
__global__ __launch_bounds__(128) void qkv_kernel(
        const bf16_t* __restrict__ xbf, const bf16_t* __restrict__ wbf,
        bf16_t* __restrict__ Qh, bf16_t* __restrict__ Kh, bf16_t* __restrict__ Vt) {
    const int lane = threadIdx.x & 31;
    const int wid  = blockIdx.x * (blockDim.x >> 5) + (threadIdx.x >> 5);
    const int NT = D_ / 64;           // 16 n-tiles
    const int MT = (B_ * L_) / 16;    // 256 m-tiles
    int mat   = wid / (MT * NT);      // 0=Q 1=K 2=V
    int rem   = wid - mat * (MT * NT);
    int mtile = rem / NT;
    int ntile = rem - mtile * NT;
    const int lr = lane & 15, half = lane >> 4;

    const bf16_t* w0 = wbf + (size_t)mat * D_ * D_;
    const int m_a = mtile * 16 + lr;                  // A-frag row (lanes 0-15 & 16-31 both M=lr)

    v8f acc[4];
#pragma unroll
    for (int f = 0; f < 4; ++f)
#pragma unroll
        for (int r = 0; r < 8; ++r) acc[f][r] = 0.0f;

    for (int kk = 0; kk < D_; kk += 32) {
        // A-frag: per-lane K = 0..15 (+16 for upper half) -> one contiguous load
        const bf16_t* ap = xbf + (size_t)m_a * D_ + kk + half * 16;
        v16bf a = *(const v16bf*)ap;
#pragma unroll
        for (int f = 0; f < 4; ++f) {
            // B-frag: B[k][n] = W[n][k]; lane holds column n, K = 0..15 (+16 for upper half)
            int n = ntile * 64 + f * 16 + lr;
            const bf16_t* bp = w0 + (size_t)n * D_ + kk + half * 16;
            v16bf b = *(const v16bf*)bp;
            acc[f] = __builtin_amdgcn_wmma_f32_16x16x32_bf16(
                false, a, false, b, (short)0, acc[f], false, false);
        }
    }

    if (mat < 2) {                       // Q / K : row-major (B,H,L,HD)
        bf16_t* dst = (mat == 0) ? Qh : Kh;
#pragma unroll
        for (int f = 0; f < 4; ++f) {
            int n = ntile * 64 + f * 16 + lr;
            int h = n >> 6, hd = n & 63;
#pragma unroll
            for (int r = 0; r < 8; ++r) {
                int m = mtile * 16 + r + half * 8;   // C layout: row = r (+8 for upper half)
                int b = m >> 11, l = m & (L_ - 1);
                dst[((size_t)(b * H_ + h) * L_ + l) * HD_ + hd] = (bf16_t)acc[f][r];
            }
        }
    } else {                             // V : transposed (B,H,HD,L) -> contiguous b128 store
#pragma unroll
        for (int f = 0; f < 4; ++f) {
            int n = ntile * 64 + f * 16 + lr;
            int h = n >> 6, hd = n & 63;
            int m0 = mtile * 16;
            int b = m0 >> 11;
            int l0 = (m0 & (L_ - 1)) + half * 8;
            v8bf pk;
#pragma unroll
            for (int r = 0; r < 8; ++r) pk[r] = (bf16_t)acc[f][r];
            *(v8bf*)(Vt + ((size_t)(b * H_ + h) * HD_ + hd) * L_ + l0) = pk;
        }
    }
}

// ---------------------------------------------------------------------------
// Kernel 3: flash attention. One wave owns 16 query rows of one (b,h);
// streams keys 32 at a time. Computes S^T = K_tile @ Q^T so the softmaxed
// probabilities are already laid out as the A-fragment for P @ V.
// ---------------------------------------------------------------------------
__global__ __launch_bounds__(128) void attn_kernel(
        const bf16_t* __restrict__ Qh, const bf16_t* __restrict__ Kh,
        const bf16_t* __restrict__ Vt, const float* __restrict__ qs,
        const float* __restrict__ scale, float* __restrict__ out) {
    const int lane = threadIdx.x & 31;
    const int wid  = blockIdx.x * (blockDim.x >> 5) + (threadIdx.x >> 5);
    const int QT = L_ / 16;           // 128 q-tiles per (b,h)
    int bh = wid / QT;
    int qt = wid - bh * QT;
    int b  = bh >> 4, h = bh & 15;
    const int lr = lane & 15, half = lane >> 4;

    const bf16_t* Q0 = Qh + (size_t)bh * L_ * HD_;
    const bf16_t* K0 = Kh + (size_t)bh * L_ * HD_;
    const bf16_t* V0 = Vt + (size_t)bh * HD_ * L_;
    const float*  S0 = qs + (size_t)b * L_ * L_;

    const float LOG2E = 1.4426950408889634f;
    float sh = scale[h];
    float c1 = 0.125f * sh * LOG2E;   // 1/sqrt(64) * scale_h, log2 domain
    float c2 = sh * LOG2E;            // bias * scale_h, log2 domain
    int qbase = qt * 16;

    // Q held as two B-fragments for the whole key loop (B[k][n] = Q[q=n][hd=k])
    v16bf qb[2];
#pragma unroll
    for (int c = 0; c < 2; ++c) {
        const bf16_t* p = Q0 + (size_t)(qbase + lr) * HD_ + c * 32 + half * 16;
        qb[c] = *(const v16bf*)p;
    }

    v8f o[4];
#pragma unroll
    for (int f = 0; f < 4; ++f)
#pragma unroll
        for (int r = 0; r < 8; ++r) o[f][r] = 0.0f;
    float m_run = -1e30f, l_run = 0.0f;

    for (int kb = 0; kb < L_; kb += 32) {
        // ---- S^T tile (32 keys x 16 q) as two C-fragments, K(contraction)=64 ----
        v8f s[2];
#pragma unroll
        for (int t = 0; t < 2; ++t) {
            v8f c0;
#pragma unroll
            for (int r = 0; r < 8; ++r) c0[r] = 0.0f;
#pragma unroll
            for (int c = 0; c < 2; ++c) {
                const bf16_t* kp = K0 + (size_t)(kb + t * 16 + lr) * HD_ + c * 32 + half * 16;
                v16bf ak = *(const v16bf*)kp;
                c0 = __builtin_amdgcn_wmma_f32_16x16x32_bf16(
                    false, ak, false, qb[c], (short)0, c0, false, false);
            }
            s[t] = c0;
        }
        if (kb + 32 < L_)
            __builtin_prefetch(K0 + (size_t)(kb + 32 + lr) * HD_, 0, 1);

        // ---- bias + per-head scale, in log2 domain ----
        // S^T C layout: lane = q (=lr), VGPR r = key (kb + t*16 + r + 8*half)
        float lg[16];
#pragma unroll
        for (int t = 0; t < 2; ++t) {
            const float* bp = S0 + (size_t)(qbase + lr) * L_ + kb + t * 16 + half * 8;
            v4f b0 = *(const v4f*)bp;
            v4f b1 = *(const v4f*)(bp + 4);
#pragma unroll
            for (int r = 0; r < 8; ++r) {
                float bias = (r < 4) ? b0[r] : b1[r - 4];
                lg[t * 8 + r] = s[t][r] * c1 + bias * c2;
            }
        }

        // ---- online softmax: per-lane reduce over 16 regs + one xor-16 shuffle ----
        float pm = lg[0];
#pragma unroll
        for (int e = 1; e < 16; ++e) pm = fmaxf(pm, lg[e]);
        pm = fmaxf(pm, __shfl_xor(pm, 16, 32));
        float m_new = fmaxf(m_run, pm);
        float alpha = exp2f(m_run - m_new);
        float p[16], psum = 0.0f;
#pragma unroll
        for (int e = 0; e < 16; ++e) { p[e] = exp2f(lg[e] - m_new); psum += p[e]; }
        psum += __shfl_xor(psum, 16, 32);
        l_run = l_run * alpha + psum;
        m_run = m_new;

        // ---- rescale O (O row = r + 8*half; alpha_q held at lane q) ----
        float av[8];
#pragma unroll
        for (int r = 0; r < 8; ++r) av[r] = __shfl(alpha, half * 8 + r, 32);
#pragma unroll
        for (int f = 0; f < 4; ++f)
#pragma unroll
            for (int r = 0; r < 8; ++r) o[f][r] *= av[r];

        // ---- P is already in A-fragment layout: just convert f32 -> bf16 ----
        v16bf pa;
#pragma unroll
        for (int e = 0; e < 16; ++e) pa[e] = (bf16_t)p[e];

        // ---- O += P @ V ; V^T gives contiguous B-fragments ----
#pragma unroll
        for (int f = 0; f < 4; ++f) {
            const bf16_t* vp = V0 + (size_t)(f * 16 + lr) * L_ + kb + half * 8;
            v16bf vb = cat8(*(const v8bf*)vp, *(const v8bf*)(vp + 16));
            o[f] = __builtin_amdgcn_wmma_f32_16x16x32_bf16(
                false, pa, false, vb, (short)0, o[f], false, false);
        }
    }

    // ---- normalize and store (B, L, D) fp32 ----
    float inv = 1.0f / l_run;
    float iv[8];
#pragma unroll
    for (int r = 0; r < 8; ++r) iv[r] = __shfl(inv, half * 8 + r, 32);
#pragma unroll
    for (int f = 0; f < 4; ++f) {
        int hd = f * 16 + lr;
#pragma unroll
        for (int r = 0; r < 8; ++r) {
            int l = qbase + r + half * 8;
            out[((size_t)b * L_ + l) * D_ + h * HD_ + hd] = o[f][r] * iv[r];
        }
    }
}

// ---------------------------------------------------------------------------
extern "C" void kernel_launch(void* const* d_in, const int* in_sizes, int n_in,
                              void* d_out, int out_size, void* d_ws, size_t ws_size,
                              hipStream_t stream) {
    const float* x  = (const float*)d_in[0];
    const float* qs = (const float*)d_in[1];
    const float* wq = (const float*)d_in[2];
    const float* wk = (const float*)d_in[3];
    const float* wv = (const float*)d_in[4];
    const float* sc = (const float*)d_in[5];
    float* out = (float*)d_out;

    bf16_t* xbf = (bf16_t*)d_ws;                       // B*L*D      = 4194304 el
    bf16_t* wbf = xbf + (size_t)B_ * L_ * D_;          // 3*D*D      = 3145728 el
    bf16_t* Qh  = wbf + (size_t)3 * D_ * D_;           // (B,H,L,HD) = 4194304 el
    bf16_t* Kh  = Qh + (size_t)B_ * L_ * D_;           // (B,H,L,HD)
    bf16_t* Vt  = Kh + (size_t)B_ * L_ * D_;           // (B,H,HD,L)

    int total4 = (B_ * L_ * D_ + 3 * D_ * D_) / 4;
    cvt_kernel<<<(total4 + 255) / 256, 256, 0, stream>>>(x, wq, wk, wv, xbf, wbf);

    int qkv_waves = 3 * (B_ * L_ / 16) * (D_ / 64);    // 12288
    qkv_kernel<<<qkv_waves / 4, 128, 0, stream>>>(xbf, wbf, Qh, Kh, Vt);

    int attn_waves = B_ * H_ * (L_ / 16);              // 4096
    attn_kernel<<<attn_waves / 4, 128, 0, stream>>>(Qh, Kh, Vt, qs, sc, out);
}
